// RVQHead_76982993814088
// MI455X (gfx1250) — matrix operand order repaired and checked by reading, not compiled, not run on hardware
//
#include <hip/hip_runtime.h>
#include <hip/hip_bf16.h>

typedef __attribute__((ext_vector_type(16))) __bf16 v16bf;
typedef __attribute__((ext_vector_type(8)))  float  v8f;

#define D_DIM 512
#define TILE_M 128           // rows per block (4 waves x 32 rows)
#define BLK_T  128           // threads per block (4 waves)
#define TILE_C 16            // codes per LDS chunk
#define BROW  (D_DIM + 8)    // LDS row stride in bf16 elems (16B pad -> conflict-free)

#if __has_builtin(__builtin_amdgcn_global_load_async_to_lds_b128) && \
    __has_builtin(__builtin_amdgcn_s_wait_asynccnt)
#define USE_ASYNC 1
#else
#define USE_ASYNC 0
#endif

// Builtin parameter types: pointer to 16-byte int vector in as(1)/as(3).
typedef int v4i_ __attribute__((vector_size(16)));
typedef __attribute__((address_space(1))) v4i_ gv4i;
typedef __attribute__((address_space(3))) v4i_ sv4i;

__device__ inline gv4i* as_global(const void* p) {
  return (gv4i*)(unsigned long long)(uintptr_t)p;
}
__device__ inline sv4i* as_shared(const void* p) {
  // generic LDS pointer low 32 bits == LDS offset
  return (sv4i*)(unsigned)(uintptr_t)p;
}

// ---------------------------------------------------------------------------
// Kernel 0: convert codebook f32 -> bf16 (workspace) and compute |c|^2 norms.
// ---------------------------------------------------------------------------
__global__ __launch_bounds__(256)
void vq_prep_codebook(const float* __restrict__ codebook,
                      __bf16* __restrict__ cbb,
                      float* __restrict__ norms, int K) {
  const int w = threadIdx.x >> 5;
  const int l = threadIdx.x & 31;
  const int k = blockIdx.x * 8 + w;
  if (k >= K) return;
  const float4* src = (const float4*)(codebook + (size_t)k * D_DIM);
  float sum = 0.f;
#pragma unroll
  for (int j = 0; j < 4; ++j) {
    const int idx = l + 32 * j;
    float4 v = src[idx];
    sum += v.x * v.x + v.y * v.y + v.z * v.z + v.w * v.w;
    union { __bf16 h[4]; uint2 u; } p;
    p.h[0] = (__bf16)v.x; p.h[1] = (__bf16)v.y;
    p.h[2] = (__bf16)v.z; p.h[3] = (__bf16)v.w;
    *((uint2*)(cbb + (size_t)k * D_DIM) + idx) = p.u;
  }
#pragma unroll
  for (int off = 16; off >= 1; off >>= 1) sum += __shfl_xor(sum, off, 32);
  if (l == 0) norms[k] = sum;
}

// Pack 16 f32 into one bf16 A-operand fragment.
__device__ inline v16bf make_frag(float4 p0, float4 p1, float4 q0, float4 q1) {
  v16bf r;
  r[0]  = (__bf16)p0.x; r[1]  = (__bf16)p0.y; r[2]  = (__bf16)p0.z; r[3]  = (__bf16)p0.w;
  r[4]  = (__bf16)p1.x; r[5]  = (__bf16)p1.y; r[6]  = (__bf16)p1.z; r[7]  = (__bf16)p1.w;
  r[8]  = (__bf16)q0.x; r[9]  = (__bf16)q0.y; r[10] = (__bf16)q0.z; r[11] = (__bf16)q0.w;
  r[12] = (__bf16)q1.x; r[13] = (__bf16)q1.y; r[14] = (__bf16)q1.z; r[15] = (__bf16)q1.w;
  return r;
}

// Load one 16-row A tile (rows base..base+15) into WMMA A-layout fragments.
__device__ inline void load_a_tile(const float* __restrict__ features,
                                   int base, int ln, int lh, v16bf a[16]) {
  const float* frow = features + (size_t)(base + ln) * D_DIM;
#pragma unroll
  for (int ks = 0; ks < 16; ++ks) {
    const int k0 = ks * 32;
    const int o1 = k0 + 16 * lh;
    const int o2 = k0 + 16 * lh + 8;
    float4 p0 = *(const float4*)(frow + o1);
    float4 p1 = *(const float4*)(frow + o1 + 4);
    float4 q0 = *(const float4*)(frow + o2);
    float4 q1 = *(const float4*)(frow + o2 + 4);
    a[ks] = make_frag(p0, p1, q0, q1);
  }
}

// ---------------------------------------------------------------------------
// Kernel 1: fused GEMM + argmin.
// Block = 128 threads = 4 waves; each wave owns TWO 16-row M-tiles (32 rows).
// Both tiles share every B fragment -> half the LDS reads per WMMA and two
// independent accumulator chains for the XDL pipe. B fragments are software-
// pipelined (bcur/bnext) so ds_load latency overlaps the WMMA pair. B tiles
// stream through double-buffered LDS via async global->LDS DMA.
// ---------------------------------------------------------------------------
__global__ __launch_bounds__(BLK_T)
void vq_argmin(const float* __restrict__ features,
               const __bf16* __restrict__ cbb,
               const float* __restrict__ norms,
               int* __restrict__ ids, int K) {
  __shared__ __bf16 sB[2][TILE_C * BROW];   // 2 x ~16.6 KB double buffer

  const int t    = threadIdx.x;
  const int w    = t >> 5;
  const int lane = t & 31;
  const int ln   = lane & 15;   // N (code) within 16-wide tile
  const int lh   = lane >> 4;   // half select
  const int r0   = blockIdx.x * TILE_M;
  const int rm   = r0 + 32 * w;

  // --- Two A tiles (32 rows x 512) -> registers.
  v16bf a0[16], a1[16];
  load_a_tile(features, rm,      ln, lh, a0);
  load_a_tile(features, rm + 16, ln, lh, a1);

  float minval0[8], minval1[8];
  int   minidx0[8], minidx1[8];
#pragma unroll
  for (int v = 0; v < 8; ++v) {
    minval0[v] = 3.4e38f; minidx0[v] = 0;
    minval1[v] = 3.4e38f; minidx1[v] = 0;
  }

  // Cooperative copy of one 16-code chunk into LDS buffer `buf`.
  // 16 rows x 64 uint4 = 1024 x b128 -> 8 per thread (128 threads).
  auto copy_chunk = [&](int buf, int cb) {
#pragma unroll
    for (int j = 0; j < 8; ++j) {
      const int i   = t + BLK_T * j;
      const int row = i >> 6;
      const int col = i & 63;
      const __bf16* g = cbb + (size_t)(cb + row) * D_DIM + col * 8;
      __bf16*       l = &sB[buf][row * BROW + col * 8];
#if USE_ASYNC
      __builtin_amdgcn_global_load_async_to_lds_b128(as_global(g), as_shared(l), 0, 0);
#else
      *(uint4*)l = *(const uint4*)g;
#endif
    }
  };

  const int nch = K / TILE_C;
  copy_chunk(0, 0);  // prefetch first chunk

  for (int c = 0; c < nch; ++c) {
    const int buf = c & 1;
    if (c + 1 < nch) {
      copy_chunk(buf ^ 1, (c + 1) * TILE_C);   // prefetch next while computing
#if USE_ASYNC
      __builtin_amdgcn_s_wait_asynccnt(8);     // this chunk's 8 loads done; prefetch in flight
#endif
    } else {
#if USE_ASYNC
      __builtin_amdgcn_s_wait_asynccnt(0);
#endif
    }
    __syncthreads();   // all waves' loads for chunk c are resident

    // 32x16 of dot(f, c): two 16x16 WMMA chains sharing each B fragment,
    // with the next fragment's ds_load in flight during the WMMA pair.
    v8f acc0 = {}, acc1 = {};
    const __bf16* bb = &sB[buf][(size_t)ln * BROW + 16 * lh];
    v16bf bcur = *(const v16bf*)(bb);
#pragma unroll
    for (int ks = 0; ks < 16; ++ks) {
      // B layout: lanes 0-15 (col=ln) hold K=k0..k0+15; lanes 16-31 K=k0+16..31.
      v16bf bnext = (ks < 15) ? *(const v16bf*)(bb + (ks + 1) * 32) : bcur;
      acc0 = __builtin_amdgcn_wmma_f32_16x16x32_bf16(
          false, a0[ks], false, bcur, (short)0, acc0, false, false);
      acc1 = __builtin_amdgcn_wmma_f32_16x16x32_bf16(
          false, a1[ks], false, bcur, (short)0, acc1, false, false);
      bcur = bnext;
    }

    // Epilogue: score = |c|^2 - 2*dot ; update per-lane running argmin.
    const int   cidx = c * TILE_C + ln;
    const float cn   = norms[cidx];
#pragma unroll
    for (int v = 0; v < 8; ++v) {
      float s0 = fmaf(acc0[v], -2.0f, cn);
      if (s0 < minval0[v]) { minval0[v] = s0; minidx0[v] = cidx; }
      float s1 = fmaf(acc1[v], -2.0f, cn);
      if (s1 < minval1[v]) { minval1[v] = s1; minidx1[v] = cidx; }
    }
    __syncthreads();   // everyone done reading buf before it is overwritten
  }

  // Cross-lane reduction within each 16-lane half (lowest-index tie-break),
  // then lane 0 of each half writes the row results directly.
#pragma unroll
  for (int v = 0; v < 8; ++v) {
    float mv0 = minval0[v]; int mi0 = minidx0[v];
    float mv1 = minval1[v]; int mi1 = minidx1[v];
#pragma unroll
    for (int off = 1; off <= 8; off <<= 1) {
      float ov0 = __shfl_xor(mv0, off, 32);
      int   oi0 = __shfl_xor(mi0, off, 32);
      if (ov0 < mv0 || (ov0 == mv0 && oi0 < mi0)) { mv0 = ov0; mi0 = oi0; }
      float ov1 = __shfl_xor(mv1, off, 32);
      int   oi1 = __shfl_xor(mi1, off, 32);
      if (ov1 < mv1 || (ov1 == mv1 && oi1 < mi1)) { mv1 = ov1; mi1 = oi1; }
    }
    if (ln == 0) {
      ids[rm + v + 8 * lh]      = mi0;  // C layout: VGPR v -> rows M=v / M=v+8
      ids[rm + 16 + v + 8 * lh] = mi1;
    }
  }
}

// ---------------------------------------------------------------------------
// Kernel 2: exact fp32 partial sums of |f - c_id|^2 (128 rows/block).
// ---------------------------------------------------------------------------
__global__ __launch_bounds__(256)
void vq_commit_partial(const float* __restrict__ features,
                       const float* __restrict__ codebook,
                       const int* __restrict__ ids,
                       float* __restrict__ partials) {
  __shared__ float sacc[8];
  const int w = threadIdx.x >> 5;
  const int l = threadIdx.x & 31;
  const int base = blockIdx.x * 128 + w * 16;
  float wsum = 0.f;
  for (int r = 0; r < 16; ++r) {
    const int row = base + r;
    const int c = ids[row];
    const float4* fp = (const float4*)(features + (size_t)row * D_DIM);
    const float4* cp = (const float4*)(codebook + (size_t)c * D_DIM);
    float s = 0.f;
#pragma unroll
    for (int j = 0; j < 4; ++j) {
      const int idx = l + 32 * j;
      float4 f = fp[idx], cc = cp[idx];
      float dx = f.x - cc.x, dy = f.y - cc.y, dz = f.z - cc.z, dw = f.w - cc.w;
      s += dx * dx + dy * dy + dz * dz + dw * dw;
    }
    wsum += s;
  }
#pragma unroll
  for (int off = 16; off >= 1; off >>= 1) wsum += __shfl_xor(wsum, off, 32);
  if (l == 0) sacc[w] = wsum;
  __syncthreads();
  if (threadIdx.x == 0) {
    float tot = 0.f;
#pragma unroll
    for (int i = 0; i < 8; ++i) tot += sacc[i];
    partials[blockIdx.x] = tot;
  }
}

__global__ __launch_bounds__(256)
void vq_commit_final(const float* __restrict__ partials,
                     float* __restrict__ out, int N) {
  __shared__ float s[256];
  s[threadIdx.x] = partials[threadIdx.x];
  __syncthreads();
  for (int off = 128; off >= 1; off >>= 1) {
    if ((int)threadIdx.x < off) s[threadIdx.x] += s[threadIdx.x + off];
    __syncthreads();
  }
  if (threadIdx.x == 0)
    out[0] = s[0] * 1.25f / ((float)N * (float)D_DIM);  // commit = 1.25 * mean
}

// ---------------------------------------------------------------------------
extern "C" void kernel_launch(void* const* d_in, const int* in_sizes, int n_in,
                              void* d_out, int out_size, void* d_ws, size_t ws_size,
                              hipStream_t stream) {
  const float* features = (const float*)d_in[0];
  const float* codebook = (const float*)d_in[1];
  const int N = in_sizes[0] / D_DIM;  // 32768
  const int K = in_sizes[1] / D_DIM;  // 8192

  char* ws = (char*)d_ws;
  __bf16* cbb   = (__bf16*)ws;                                       // 8 MB
  float*  norms = (float*)(ws + (size_t)K * D_DIM * sizeof(__bf16)); // 32 KB
  float*  partials = norms + K;                                      // 1 KB

  int*   ids    = (int*)d_out;          // int32 bits for the (B,T) indices
  float* commit = (float*)d_out + N;    // scalar loss

  hipLaunchKernelGGL(vq_prep_codebook, dim3(K / 8), dim3(256), 0, stream,
                     codebook, cbb, norms, K);
  hipLaunchKernelGGL(vq_argmin, dim3(N / TILE_M), dim3(BLK_T), 0, stream,
                     features, cbb, norms, ids, K);
  hipLaunchKernelGGL(vq_commit_partial, dim3(N / 128), dim3(256), 0, stream,
                     features, codebook, ids, partials);
  hipLaunchKernelGGL(vq_commit_final, dim3(1), dim3(256), 0, stream,
                     partials, commit, N);
}
